// OccNet_5841155522592
// MI455X (gfx1250) — compile-verified, matrix operand order checked by reading, not run on hardware
//
#include <hip/hip_runtime.h>
#include <hip/hip_bf16.h>

typedef __attribute__((ext_vector_type(16))) __bf16 v16bf;
typedef __attribute__((ext_vector_type(8)))  __bf16 v8bf;
typedef __attribute__((ext_vector_type(8)))  float  v8f;

#define STR 520            // padded LDS row stride in bf16 elems (1040B: 16B-aligned, +4 banks/row)
#define H1_OFF   0         // 128*520*2 = 133120 B
#define W2_OFF   133120    // 64*520*2  = 66560  B
#define BC2_OFF  199680    // 512*4     = 2048   B
#define WC3_OFF  201728    // 1536*4    = 6144   B
#define FACC_OFF 207872    // 384*4     = 1536   B
#define LDS_BYTES 209408

// ---------------- small prologue kernels (<0.1% of FLOPs) ----------------

__global__ __launch_bounds__(256) void k_lin1(const float* __restrict__ fea,
                                              const float* __restrict__ W,
                                              const float* __restrict__ bias,
                                              float* __restrict__ y1) {
  int i = blockIdx.x * 256 + threadIdx.x;        // 8192 outputs (8 x 1024)
  int b = i >> 10, j = i & 1023;
  float acc = bias[j];
  const float* fr = fea + b * 1024;
  for (int k = 0; k < 1024; ++k) acc += fr[k] * W[k * 1024 + j];
  y1[i] = acc > 0.f ? acc : 0.f;
}

__global__ __launch_bounds__(256) void k_lin2(const float* __restrict__ y1,
                                              const float* __restrict__ W,
                                              const float* __restrict__ bias,
                                              float* __restrict__ out_x,
                                              float* __restrict__ xw) {
  int i = blockIdx.x * 256 + threadIdx.x;        // 24576 outputs (8 x 3072)
  int b = i / 3072, j = i % 3072;
  float acc = bias[j];
  const float* yr = y1 + b * 1024;
  for (int k = 0; k < 1024; ++k) acc += yr[k] * W[k * 3072 + j];
  out_x[i] = acc;                                 // coarse output (B, 1024, 3) flat
  xw[i]    = acc;                                 // kept for fine stage
}

// base[b][j] = fea[b] . W_c1[5:,j] + b_c1[j]   (shared by all 16384 fine pts of batch b)
__global__ __launch_bounds__(256) void k_base(const float* __restrict__ fea,
                                              const float* __restrict__ Wc1,
                                              const float* __restrict__ bc1,
                                              float* __restrict__ basep) {
  int i = blockIdx.x * 256 + threadIdx.x;        // 4096 outputs (8 x 512)
  int b = i >> 9, j = i & 511;
  float acc = bc1[j];
  const float* fr = fea + b * 1024;
  const float* w  = Wc1 + 5 * 512 + j;
  for (int k = 0; k < 1024; ++k) acc += fr[k] * w[k * 512];
  basep[i] = acc;
}

// gridtab[p][j] = gx(p)*W_c1[0,j] + gy(p)*W_c1[1,j]   (only 16 distinct grid rows)
__global__ __launch_bounds__(256) void k_grid(const float* __restrict__ Wc1,
                                              float* __restrict__ gridtab) {
  int i = blockIdx.x * 256 + threadIdx.x;        // 8192 outputs (16 x 512)
  int p = i >> 9, j = i & 511;
  const float step = 0.1f / 3.f;
  float gx = -0.05f + step * (float)(p & 3);
  float gy = -0.05f + step * (float)(p >> 2);
  gridtab[i] = gx * Wc1[j] + gy * Wc1[512 + j];
}

// W2t[n][k] = bf16(W_c2[k][n]) : one-time transpose+downconvert; 512KB stays hot in L2
__global__ __launch_bounds__(256) void k_w2t(const float* __restrict__ Wc2,
                                             __bf16* __restrict__ W2t) {
  int i = blockIdx.x * 256 + threadIdx.x;        // 262144
  int k = i >> 9, n = i & 511;                   // coalesced reads of W_c2 rows
  W2t[n * 512 + k] = (__bf16)Wc2[k * 512 + n];
}

// ---------------- fused fine-point kernel (WMMA bf16) ----------------
// One block = 128 fine rows (8 waves, one 16-row M-tile each). h1 (128x512) is
// built in LDS in bf16, layer-2 runs on v_wmma_f32_16x16x32_bf16 against
// pre-transposed bf16 W_c2 chunks staged into LDS, layer-3 (512->3) is fused
// into the epilogue via per-lane partials + ds_add_f32 reduction.

__global__ __launch_bounds__(256) void k_fine(
    const float* __restrict__ basep,     // [8][512]  (includes b_c1)
    const float* __restrict__ gridtab,   // [16][512]
    const float* __restrict__ xcoarse,   // [8][3072]
    const float* __restrict__ Wc1,       // rows 2..4 used (point part)
    const __bf16* __restrict__ W2t,      // [512][512] n-major, k fast (bf16)
    const float* __restrict__ bc2,       // [512]
    const float* __restrict__ Wc3,       // [512][3]
    const float* __restrict__ bc3,       // [3]
    float* __restrict__ fine)            // [131072][3]
{
  extern __shared__ char lds[];
  __bf16* h1s  = (__bf16*)(lds + H1_OFF);    // [128][STR]
  __bf16* w2s  = (__bf16*)(lds + W2_OFF);    // [64][STR]  (n-major, k fast)
  float*  bc2s = (float*)(lds + BC2_OFF);
  float*  wc3s = (float*)(lds + WC3_OFF);
  float*  facc = (float*)(lds + FACC_OFF);   // [128][3]

  const int tid = threadIdx.x;
  const int R0  = blockIdx.x * 128;          // 128 | 16384 -> no batch crossing
  const int b   = R0 >> 14;
  const int f0  = R0 & 16383;

  // preload biases / Wc3, zero the layer-3 accumulator
  for (int i = tid; i < 512;  i += 256) bc2s[i] = bc2[i];
  for (int i = tid; i < 1536; i += 256) wc3s[i] = Wc3[i];
  for (int i = tid; i < 384;  i += 256) facc[i] = 0.f;

  // ---- build h1 tile: h1[r][j] = relu(base + grid + x3 . Wc1[2:5,j]) ----
  const float* w2r = Wc1 + 2 * 512;
  const float* w3r = Wc1 + 3 * 512;
  const float* w4r = Wc1 + 4 * 512;
  for (int e = tid; e < 128 * 512; e += 256) {
    int r = e >> 9, j = e & 511;
    int f = f0 + r;
    int c = f >> 4, p = f & 15;
    const float* x3 = xcoarse + b * 3072 + c * 3;
    float v = basep[b * 512 + j] + gridtab[p * 512 + j]
            + x3[0] * w2r[j] + x3[1] * w3r[j] + x3[2] * w4r[j];
    v = v > 0.f ? v : 0.f;
    h1s[r * STR + j] = (__bf16)v;
  }
  __syncthreads();

  const int wave = tid >> 5;                 // wave -> 16-row M-tile (0..7)
  const int lane = tid & 31;
  const int lm   = lane & 15;
  const int half = lane >> 4;

  float p3[8][3];
#pragma unroll
  for (int r = 0; r < 8; ++r) { p3[r][0] = 0.f; p3[r][1] = 0.f; p3[r][2] = 0.f; }

  for (int ch = 0; ch < 8; ++ch) {           // 8 N-chunks of 64 columns
    __syncthreads();
    // stage pre-transposed bf16 W2t chunk: 64 rows x 512 k = 64KB, pure memcpy
    {
      // 4096 uint4 total; thread t moves 16 of them, coalesced per LDS row
      for (int idx = tid; idx < 4096; idx += 256) {
        int n = idx >> 6;                    // 64 uint4 per 1KB row
        int q = idx & 63;
        const uint4* src = (const uint4*)(W2t + (ch * 64 + n) * 512);
        uint4*       dst = (uint4*)(w2s + n * STR);
        dst[q] = src[q];
      }
    }
    // prefetch next chunk into L2/WGP$ while we compute this one
    if (ch + 1 < 8) {
      int n   = tid >> 2;                    // 64 rows, 4 lines of 256B each
      int off = (tid & 3) * 128;             // 128 bf16 = 256B
      __builtin_prefetch((const void*)(W2t + (ch * 64 + 64 + n) * 512 + off), 0, 3);
    }
    __syncthreads();

    v8f acc[4] = {};                         // 4 N-tiles x 16x16 f32
#pragma unroll 4
    for (int kb = 0; kb < 16; ++kb) {        // K = 512 in steps of 32
      // A fragment: lane holds row m=lm, K runs [half*8,+8) and [16+half*8,+8)
      const __bf16* arow = h1s + (wave * 16 + lm) * STR + kb * 32 + half * 8;
      v8bf alo = *(const v8bf*)(arow);
      v8bf ahi = *(const v8bf*)(arow + 16);
      v16bf afrag = __builtin_shufflevector(alo, ahi,
          0,1,2,3,4,5,6,7,8,9,10,11,12,13,14,15);
#pragma unroll
      for (int nt = 0; nt < 4; ++nt) {
        // B fragment: lane holds col n = nt*16+lm, same K pattern (w2s is n-major)
        const __bf16* brow = w2s + (nt * 16 + lm) * STR + kb * 32 + half * 8;
        v8bf blo = *(const v8bf*)(brow);
        v8bf bhi = *(const v8bf*)(brow + 16);
        v16bf bfrag = __builtin_shufflevector(blo, bhi,
            0,1,2,3,4,5,6,7,8,9,10,11,12,13,14,15);
        acc[nt] = __builtin_amdgcn_wmma_f32_16x16x32_bf16(
            /*neg_a=*/false, afrag, /*neg_b=*/false, bfrag,
            /*c_mod=*/(short)0, acc[nt], /*reuse_a=*/false, /*reuse_b=*/false);
      }
    }

    // epilogue for this chunk: h2 = relu(acc + b_c2); fold layer-3 partials
#pragma unroll
    for (int nt = 0; nt < 4; ++nt) {
      int n = ch * 64 + nt * 16 + lm;
      float bias = bc2s[n];
      float w0 = wc3s[n * 3 + 0], w1 = wc3s[n * 3 + 1], w2 = wc3s[n * 3 + 2];
#pragma unroll
      for (int r = 0; r < 8; ++r) {          // C layout: m = r + 8*half
        float h = acc[nt][r] + bias;
        h = h > 0.f ? h : 0.f;
        p3[r][0] += h * w0;
        p3[r][1] += h * w1;
        p3[r][2] += h * w2;
      }
    }
  }

  // cross-lane (over n) reduction of layer-3 partials
#pragma unroll
  for (int r = 0; r < 8; ++r) {
    int m = wave * 16 + half * 8 + r;
    atomicAdd(&facc[m * 3 + 0], p3[r][0]);
    atomicAdd(&facc[m * 3 + 1], p3[r][1]);
    atomicAdd(&facc[m * 3 + 2], p3[r][2]);
  }
  __syncthreads();

  // fine = facc + b_c3 + point_feat (center)
  for (int e = tid; e < 384; e += 256) {
    int r = e / 3, coord = e % 3;
    int f = f0 + r, c = f >> 4;
    fine[(R0 + r) * 3 + coord] =
        facc[e] + bc3[coord] + xcoarse[b * 3072 + c * 3 + coord];
  }
}

// ---------------- host side ----------------

extern "C" void kernel_launch(void* const* d_in, const int* in_sizes, int n_in,
                              void* d_out, int out_size, void* d_ws, size_t ws_size,
                              hipStream_t stream) {
  const float* fea  = (const float*)d_in[0];
  const float* W_l1 = (const float*)d_in[1];
  const float* b_l1 = (const float*)d_in[2];
  const float* W_l2 = (const float*)d_in[3];
  const float* b_l2 = (const float*)d_in[4];
  const float* W_c1 = (const float*)d_in[5];
  const float* b_c1 = (const float*)d_in[6];
  const float* W_c2 = (const float*)d_in[7];
  const float* b_c2 = (const float*)d_in[8];
  const float* W_c3 = (const float*)d_in[9];
  const float* b_c3 = (const float*)d_in[10];

  float* out_x    = (float*)d_out;          // 8*1024*3 = 24576
  float* out_fine = out_x + 24576;          // 8*16384*3 = 393216

  float*  y1      = (float*)d_ws;           // 8*1024
  float*  xw      = y1 + 8 * 1024;          // 8*3072
  float*  basep   = xw + 8 * 3072;          // 8*512
  float*  gridtab = basep + 8 * 512;        // 16*512
  __bf16* W2t     = (__bf16*)(gridtab + 16 * 512); // 512*512 bf16 = 512KB

  k_lin1<<<32, 256, 0, stream>>>(fea, W_l1, b_l1, y1);
  k_lin2<<<96, 256, 0, stream>>>(y1, W_l2, b_l2, out_x, xw);
  k_base<<<16, 256, 0, stream>>>(fea, W_c1, b_c1, basep);
  k_grid<<<32, 256, 0, stream>>>(W_c1, gridtab);
  k_w2t<<<1024, 256, 0, stream>>>(W_c2, W2t);
  k_fine<<<1024, 256, LDS_BYTES, stream>>>(basep, gridtab, xw, W_c1,
                                           W2t, b_c2, W_c3, b_c3, out_fine);
}
